// SelfAttention_10934986735933
// MI455X (gfx1250) — matrix operand order repaired and checked by reading, not compiled, not run on hardware
//
#include <hip/hip_runtime.h>

typedef __attribute__((ext_vector_type(16))) _Float16 v16h;
typedef __attribute__((ext_vector_type(8)))  _Float16 v8h;
typedef __attribute__((ext_vector_type(8)))  float    v8f;
typedef __attribute__((ext_vector_type(4)))  float    v4f;

#define B_   8
#define D_   512
#define L_   2048
#define H_   8
#define DH_  64
#define NEGINF  (-3.0e38f)
#define NEGMASK (-1.0e30f)
#define PADR 36   // floats per slab row: 32 data + 4 pad (144B = 9*16B)

static __device__ __forceinline__ v16h combine16(v8h lo, v8h hi) {
  v16h r;
#pragma unroll
  for (int i = 0; i < 8; ++i) { r[i] = lo[i]; r[i + 8] = hi[i]; }
  return r;
}

static __device__ __forceinline__ v8f wmma_f16(v16h a, v16h b, v8f c) {
  // D = A(16x32 f16) * B(32x16 f16) + C(16x16 f32)
  return __builtin_amdgcn_wmma_f32_16x16x32_f16(false, a, false, b,
                                                (short)0, c, false, false);
}

// ---------------------------------------------------------------------------
// Kernel 1: fused projection + precision cast.
//   Qh[b,h,l,dh] = (w_q  [o,:] . X[:,l]) * DH^-0.5   (o = h*64+dh)
//   Kh[b,h,l,dh] =  w_mem[o,:]     . X[:,l]
//   Vt[b,h,dh,l] =  w_mem[o+512,:] . X[:,l]          (transposed for PV B-frags)
// 8 waves/block share a transposed 32x128 fp32 activation slab filled with
// global_load_async_to_lds_b32; B-frags read back as contiguous ds_load_b128.
// ---------------------------------------------------------------------------
__global__ __launch_bounds__(256)
void proj_kernel(const float* __restrict__ q_in,   // (B, D, L) fp32
                 const float* __restrict__ w_mem,  // (2D, D)  fp32
                 const float* __restrict__ w_q,    // (D, D)   fp32
                 _Float16* __restrict__ Qh,        // (B,H,L,DH) f16
                 _Float16* __restrict__ Kh,        // (B,H,L,DH) f16
                 _Float16* __restrict__ Vt)        // (B,H,DH,L) f16
{
  __shared__ __align__(16) float slabT[128 * PADR]; // [col l][row k], padded

  const int tid  = threadIdx.x;
  const int lane = tid & 31;
  const int wv   = tid >> 5;
  const int m    = lane & 15;   // A-frag row / C-frag column
  const int hi   = lane >> 4;   // half-wave select
  const int sel  = blockIdx.y;  // 0=Q, 1=K, 2=V
  const int b    = blockIdx.z;

  const int o_base = ((blockIdx.x * 8) >> 7) * 16;  // same for whole block
  const int l0     = ((blockIdx.x * 8) & 127) * 16; // block l origin
  const int l_base = l0 + wv * 16;                  // wave l origin

  const int row = o_base + m;                       // weight row for A-frag
  const float* wrow =
      (sel == 0) ? (w_q   + (size_t)row * D_)
    : (sel == 1) ? (w_mem + (size_t)row * D_)
                 : (w_mem + (size_t)(row + D_) * D_);
  const float* X = q_in + (size_t)b * D_ * L_;

  v8f acc = {};
  for (int kk = 0; kk < D_; kk += 32) {
    __syncthreads();  // previous-iteration slab reads complete (all waves)

    // ---- async transpose-fill: slabT[c][r] = X[kk+r][l0+c] ----
    // consecutive lanes -> consecutive c: coalesced global, scattered LDS.
#pragma unroll
    for (int j = 0; j < 16; ++j) {
      const int idx = tid + j * 256;       // 4096 = 32 rows x 128 cols
      const int r   = idx >> 7;
      const int c   = idx & 127;
      const float*   gaddr = X + (size_t)(kk + r) * L_ + (l0 + c);
      const unsigned laddr = (unsigned)(uintptr_t)(slabT + c * PADR + r);
      asm volatile("global_load_async_to_lds_b32 %0, %1, off"
                   :: "v"(laddr), "v"(gaddr) : "memory");
    }
    asm volatile("s_wait_asynccnt 0x0" ::: "memory");
    __syncthreads();  // slab visible to all waves

    // ---- A fragment from weights: slot i <- K=hi*16+i ----
    const float* a0p = wrow + kk + hi * 16;
    const v4f wa0 = *(const v4f*)(a0p);
    const v4f wa1 = *(const v4f*)(a0p + 4);
    const v4f wb0 = *(const v4f*)(a0p + 8);
    const v4f wb1 = *(const v4f*)(a0p + 12);
    v16h a;
#pragma unroll
    for (int i = 0; i < 4; ++i) {
      a[i]      = (_Float16)wa0[i];
      a[i + 4]  = (_Float16)wa1[i];
      a[i + 8]  = (_Float16)wb0[i];
      a[i + 12] = (_Float16)wb1[i];
    }

    // ---- B fragment: slot i <- slabT[wv*16+m][hi*16+i] (64B contiguous) ----
    const float* srow = slabT + (wv * 16 + m) * PADR + hi * 16;
    v16h bf;
#pragma unroll
    for (int i = 0; i < 16; ++i)
      bf[i] = (_Float16)srow[i];

    acc = wmma_f16(a, bf, acc);
  }

  // ---- C-tile store: lane column = m, rows dh0..dh0+7 (one head per tile) ----
  const int l   = l_base + m;
  const int hh  = o_base >> 6;                 // constant per tile
  const int dh0 = (o_base & 63) + 8 * hi;      // 16B-aligned
  if (sel == 2) {
#pragma unroll
    for (int r = 0; r < 8; ++r)
      Vt[(((size_t)b * H_ + hh) * DH_ + dh0 + r) * L_ + l] = (_Float16)acc[r];
  } else {
    const float s = (sel == 0) ? 0.125f : 1.0f;   // DH^-0.5 folded into Q
    v8h ov;
#pragma unroll
    for (int r = 0; r < 8; ++r) ov[r] = (_Float16)(acc[r] * s);
    _Float16* dst = ((sel == 0) ? Qh : Kh) +
                    (((size_t)b * H_ + hh) * L_ + l) * DH_ + dh0;
    *(v8h*)dst = ov;   // single global_store_b128
  }
}

// ---------------------------------------------------------------------------
// Kernel 2: flash attention. One wave = 16 query rows; streams 32-key blocks:
// S = Q K^T (4 wmma), mask, online softmax (shfl butterflies + v_exp_f32),
// P staged through per-wave LDS (C-layout -> A-layout), O += P V (4 wmma).
// ---------------------------------------------------------------------------
__global__ __launch_bounds__(256)
void attn_kernel(const _Float16* __restrict__ Qh,   // (B,H,L,DH)
                 const _Float16* __restrict__ Kh,   // (B,H,L,DH)
                 const _Float16* __restrict__ Vt,   // (B,H,DH,L)
                 const int* __restrict__ mask,      // (B,L)
                 float* __restrict__ out)           // (B,D,L) fp32
{
  __shared__ __align__(16) _Float16 pstage[8][16 * 32]; // 1KB per wave

  const int lane = threadIdx.x & 31;
  const int wv   = threadIdx.x >> 5;
  const int n    = lane & 15;
  const int hi   = lane >> 4;
  const int h    = blockIdx.y;
  const int b    = blockIdx.z;
  const int q_base = (blockIdx.x * 8 + wv) * 16;

  const _Float16* Qbh  = Qh + (size_t)(b * H_ + h) * L_ * DH_;
  const _Float16* Kbh  = Kh + (size_t)(b * H_ + h) * L_ * DH_;
  const _Float16* Vbh  = Vt + (size_t)(b * H_ + h) * DH_ * L_;
  const int*      mrow = mask + (size_t)b * L_;

  // Q A-fragments (row = n), K-dim split 0..31 / 32..63
  const _Float16* qrow = Qbh + (size_t)(q_base + n) * DH_;
  const v16h aq0 = combine16(*(const v8h*)(qrow + hi * 16),
                             *(const v8h*)(qrow + hi * 16 + 8));
  const v16h aq1 = combine16(*(const v8h*)(qrow + 32 + hi * 16),
                             *(const v8h*)(qrow + 32 + hi * 16 + 8));

  v8f oacc[4];
#pragma unroll
  for (int f = 0; f < 4; ++f) oacc[f] = (v8f){};
  float mrun[8], lrun[8];
#pragma unroll
  for (int r = 0; r < 8; ++r) { mrun[r] = NEGINF; lrun[r] = 0.0f; }

  _Float16* pb = pstage[wv];

  for (int kb = 0; kb < L_; kb += 32) {
    // hint next key block into the caches
    __builtin_prefetch(Kbh + (size_t)(kb + 32 + n) * DH_, 0, 0);

    // ---- S = Q K^T over 32 keys (two 16x16 tiles) ----
    const _Float16* k0 = Kbh + (size_t)(kb + n) * DH_;
    const _Float16* k1 = Kbh + (size_t)(kb + 16 + n) * DH_;
    const v16h b00 = combine16(*(const v8h*)(k0 + hi * 16),
                               *(const v8h*)(k0 + hi * 16 + 8));
    const v16h b10 = combine16(*(const v8h*)(k0 + 32 + hi * 16),
                               *(const v8h*)(k0 + 32 + hi * 16 + 8));
    const v16h b01 = combine16(*(const v8h*)(k1 + hi * 16),
                               *(const v8h*)(k1 + hi * 16 + 8));
    const v16h b11 = combine16(*(const v8h*)(k1 + 32 + hi * 16),
                               *(const v8h*)(k1 + 32 + hi * 16 + 8));
    v8f s0 = {}, s1 = {};
    s0 = wmma_f16(aq0, b00, s0);
    s0 = wmma_f16(aq1, b10, s0);
    s1 = wmma_f16(aq0, b01, s1);
    s1 = wmma_f16(aq1, b11, s1);

    // ---- mask: logits*m + (1-m)*NEG ----
    const int mv0 = mrow[kb + n];
    const int mv1 = mrow[kb + 16 + n];
#pragma unroll
    for (int r = 0; r < 8; ++r) {
      if (!mv0) s0[r] = NEGMASK;
      if (!mv1) s1[r] = NEGMASK;
    }

    // ---- online softmax (rows live across the 16-lane group) ----
    float pr0[8], pr1[8], alpha[8];
#pragma unroll
    for (int r = 0; r < 8; ++r) {
      float rm = fmaxf(s0[r], s1[r]);
#pragma unroll
      for (int off = 8; off >= 1; off >>= 1)
        rm = fmaxf(rm, __shfl_xor(rm, off, 32));
      const float mnew = fmaxf(mrun[r], rm);
      alpha[r] = __expf(mrun[r] - mnew);
      mrun[r]  = mnew;
      const float p0 = __expf(s0[r] - mnew);
      const float p1 = __expf(s1[r] - mnew);
      pr0[r] = p0; pr1[r] = p1;
      float rs = p0 + p1;
#pragma unroll
      for (int off = 8; off >= 1; off >>= 1)
        rs += __shfl_xor(rs, off, 32);
      lrun[r] = lrun[r] * alpha[r] + rs;
    }
#pragma unroll
    for (int r = 0; r < 8; ++r) {
      oacc[0][r] *= alpha[r]; oacc[1][r] *= alpha[r];
      oacc[2][r] *= alpha[r]; oacc[3][r] *= alpha[r];
    }

    // ---- P: C-layout -> LDS -> A-layout (same-wave LDS ops are in-order) ----
#pragma unroll
    for (int r = 0; r < 8; ++r) {
      pb[(r + 8 * hi) * 32 + n]      = (_Float16)pr0[r];
      pb[(r + 8 * hi) * 32 + 16 + n] = (_Float16)pr1[r];
    }
    const _Float16* prow = pb + n * 32;
    const v16h pa = combine16(*(const v8h*)(prow + hi * 16),
                              *(const v8h*)(prow + hi * 16 + 8));

    // ---- O += P V  (B-frags contiguous thanks to transposed V) ----
#pragma unroll
    for (int f = 0; f < 4; ++f) {
      const _Float16* vf = Vbh + (size_t)(f * 16 + n) * L_ + kb;
      const v16h bv = combine16(*(const v8h*)(vf + hi * 16),
                                *(const v8h*)(vf + hi * 16 + 8));
      oacc[f] = wmma_f16(pa, bv, oacc[f]);
    }
  }

  // ---- normalize + store out[b, h*64+dh, l] ----
#pragma unroll
  for (int r = 0; r < 8; ++r) {
    const float inv = 1.0f / lrun[r];
    const int l = q_base + r + 8 * hi;
#pragma unroll
    for (int f = 0; f < 4; ++f)
      out[((size_t)b * D_ + h * DH_ + f * 16 + n) * L_ + l] = oacc[f][r] * inv;
  }
}

// ---------------------------------------------------------------------------
extern "C" void kernel_launch(void* const* d_in, const int* in_sizes, int n_in,
                              void* d_out, int out_size, void* d_ws, size_t ws_size,
                              hipStream_t stream) {
  const float* queries = (const float*)d_in[0];   // (B,D,L) fp32
  const int*   mask    = (const int*)  d_in[1];   // (B,L)   int32
  const float* w_mem   = (const float*)d_in[2];   // (2D,D)  fp32
  const float* w_q     = (const float*)d_in[3];   // (D,D)   fp32
  float* out = (float*)d_out;                     // (B,D,L) fp32

  const size_t elems = (size_t)B_ * H_ * L_ * DH_; // 8.4M halfs per tensor
  _Float16* Qh = (_Float16*)d_ws;
  _Float16* Kh = Qh + elems;
  _Float16* Vt = Kh + elems;

  dim3 pgrid(512, 3, B_);      // 4096 16x16 tiles per (b,sel), 8 waves/block
  proj_kernel<<<pgrid, 256, 0, stream>>>(queries, w_mem, w_q, Qh, Kh, Vt);

  dim3 agrid(L_ / 128, H_, B_); // 8 waves/block, 16 q-rows/wave
  attn_kernel<<<agrid, 256, 0, stream>>>(Qh, Kh, Vt, mask, out);
}
